// GAT_79224966742097
// MI455X (gfx1250) — matrix operand order repaired and checked by reading, not compiled, not run on hardware
//
#include <hip/hip_runtime.h>
#include <hip/hip_bf16.h>
#include <cmath>

// ---------------------------------------------------------------------------
// GAT on MI455X (gfx1250): wave32, WMMA f32<-f16 16x16x32.
//  * s + d == hp @ (a_src + a_dst) -> one logits GEMM per layer.
//  * All B operands pre-swizzled to fragment-major layout (one v16h load/lane).
//  * A fragments = two contiguous 16B runs per lane -> b128 loads.
//  * Flash-style 16-row strips, logits f16 in LDS, softmax normalization
//    deferred into the epilogue scale.
// ---------------------------------------------------------------------------

typedef _Float16 v16h __attribute__((ext_vector_type(16)));
typedef _Float16 v8h  __attribute__((ext_vector_type(8)));
typedef float    v8f  __attribute__((ext_vector_type(8)));

__device__ __forceinline__ v8f wmma_f16(v16h a, v16h b, v8f c) {
    return __builtin_amdgcn_wmma_f32_16x16x32_f16(
        false, a, false, b, (short)0, c, false, false);
}

// A fragment (16x32 f16) from row-major src (leading dim ld, halves), pre-offset
// to tile origin. Lane layout: M = lane&15; K halves {kb..kb+15},
// kb = (lane>=16)*16 -> two contiguous 16B runs => b128 loads.
__device__ __forceinline__ v16h loadA(const _Float16* __restrict__ src,
                                      int ld, int lane) {
    const int M  = lane & 15;
    const int kb = (lane >> 4) << 4;
    const _Float16* r = src + (size_t)M * ld + kb;
    v8h lo = *(const v8h*)(r);
    v8h hi = *(const v8h*)(r + 8);
    return __builtin_shufflevector(lo, hi, 0, 1, 2, 3, 4, 5, 6, 7,
                                   8, 9, 10, 11, 12, 13, 14, 15);
}

// B fragment from fragment-major packed storage: frag = 512 halves, lane's 16
// contiguous halves at frag + lane*16  => one 32B load.
__device__ __forceinline__ v16h loadBpk(const _Float16* __restrict__ frag,
                                        int lane) {
    return ((const v16h*)frag)[lane];
}

// ---------------------------------------------------------------------------
// Pack kernels: one wave per 32x16 B fragment. Strided gather (once per
// matrix), contiguous 32B store per lane.
// B layout rule: b[i] = SRC[k0 + kb + i][n0 + (lane&15)], kb = (lane>=16)*16.
// ---------------------------------------------------------------------------
__global__ void pack_asum_kernel(const float* __restrict__ a,
                                 const float* __restrict__ b,
                                 _Float16* __restrict__ dst, int H, int N) {
    const int w    = (blockIdx.x * blockDim.x + threadIdx.x) >> 5;
    const int lane = threadIdx.x & 31;
    const int nfrag = H * (N / 16) * 2;
    if (w >= nfrag) return;
    const int kc = w & 1;
    const int jt = (w >> 1) % (N / 16);
    const int h  = (w >> 1) / (N / 16);
    const int nc = lane & 15, kb = (lane >> 4) << 4;
    const int n  = jt * 16 + nc;
    v16h v;
#pragma unroll
    for (int i = 0; i < 16; ++i) {
        const size_t idx = ((size_t)h * 64 + kc * 32 + kb + i) * N + n;
        v[i] = (_Float16)(a[idx] + b[idx]);
    }
    ((v16h*)dst)[(size_t)w * 32 + lane] = v;
}

__global__ void pack_w_kernel(const float* __restrict__ w,
                              _Float16* __restrict__ dst, int H, int K) {
    const int f    = (blockIdx.x * blockDim.x + threadIdx.x) >> 5;
    const int lane = threadIdx.x & 31;
    const int nfrag = H * (K / 32) * 4;
    if (f >= nfrag) return;
    const int t  = f & 3;
    const int kt = (f >> 2) % (K / 32);
    const int h  = (f >> 2) / (K / 32);
    const int nc = lane & 15, kb = (lane >> 4) << 4;
    v16h v;
#pragma unroll
    for (int i = 0; i < 16; ++i)
        v[i] = (_Float16)w[((size_t)h * K + kt * 32 + kb + i) * 64 + t * 16 + nc];
    ((v16h*)dst)[(size_t)f * 32 + lane] = v;
}

__global__ void pack_hp_kernel(const _Float16* __restrict__ hp,
                               _Float16* __restrict__ dst, int BH, int N) {
    const int f    = (blockIdx.x * blockDim.x + threadIdx.x) >> 5;
    const int lane = threadIdx.x & 31;
    const int nfrag = BH * (N / 32) * 4;
    if (f >= nfrag) return;
    const int t  = f & 3;
    const int kt = (f >> 2) % (N / 32);
    const int bh = (f >> 2) / (N / 32);
    const int nc = lane & 15, kb = (lane >> 4) << 4;
    v16h v;
#pragma unroll
    for (int i = 0; i < 16; ++i)
        v[i] = hp[((size_t)bh * N + kt * 32 + kb + i) * 64 + t * 16 + nc];
    ((v16h*)dst)[(size_t)f * 32 + lane] = v;
}

// ---------------------------------------------------------------------------
// Instance norm over node axis: x[b][n][f] -> f16. grid.x = batch, block 1024.
// ---------------------------------------------------------------------------
__global__ __launch_bounds__(1024)
void inorm_kernel(const float* __restrict__ x, _Float16* __restrict__ o,
                  int N, int F) {
    __shared__ float sred[2048];
    const int b = blockIdx.x, tid = threadIdx.x;
    const int G = blockDim.x / F, f = tid % F, g = tid / F;
    const size_t base = (size_t)b * N * F;

    float s = 0.f, q = 0.f;
    for (int r = g; r < N; r += G) {
        float v = x[base + (size_t)r * F + f];
        s += v; q += v * v;
    }
    sred[tid] = s; sred[1024 + tid] = q;
    for (int st = G >> 1; st > 0; st >>= 1) {
        __syncthreads();
        if (g < st) {
            sred[tid]        += sred[tid + st * F];
            sred[1024 + tid] += sred[1024 + tid + st * F];
        }
    }
    __syncthreads();
    const float invN = 1.0f / (float)N;
    const float mean = sred[f] * invN;
    const float var  = sred[1024 + f] * invN - mean * mean;   // biased
    const float inv  = rsqrtf(var + 1e-5f);
    for (int r = g; r < N; r += G) {
        size_t idx = base + (size_t)r * F + f;
        o[idx] = (_Float16)((x[idx] - mean) * inv);
    }
}

// ---------------------------------------------------------------------------
// hp[b,h] = Xn[b] (NxK f16) @ W[h] (Kx64, fragment-packed) -> f16 [BH][N][64]
// grid = (B*H, N/128); block 256 (8 waves x 16-row tiles). No LDS needed.
// ---------------------------------------------------------------------------
__global__ __launch_bounds__(256)
void hpgemm_kernel(const _Float16* __restrict__ X,
                   const _Float16* __restrict__ WPK,
                   _Float16* __restrict__ HP, int N, int K, int H) {
    const int bh = blockIdx.x, h = bh % H, b = bh / H;
    const int tid = threadIdx.x, wave = tid >> 5, lane = tid & 31;
    const int r0 = blockIdx.y * 128 + wave * 16;
    const _Float16* Xb = X + ((size_t)b * N + r0) * K;

    v8f acc[4] = {};
    for (int kt = 0; kt < K; kt += 32) {
        v16h a = loadA(Xb + kt, K, lane);
        const _Float16* fb = WPK + (((size_t)h * (K / 32) + (kt >> 5)) * 4) * 512;
#pragma unroll
        for (int t = 0; t < 4; ++t)
            acc[t] = wmma_f16(a, loadBpk(fb + t * 512, lane), acc[t]);
    }

    _Float16* dst = HP + ((size_t)bh * N + r0) * 64;
    const int rowoff = (lane >> 4) << 3, nc = lane & 15;
#pragma unroll
    for (int t = 0; t < 4; ++t)
#pragma unroll
        for (int r = 0; r < 8; ++r)
            dst[(size_t)(r + rowoff) * 64 + t * 16 + nc] = (_Float16)acc[t][r];
}

// ---------------------------------------------------------------------------
// Attention strip: one WG per (b*h, 16 rows). Logits (f16, 16xN) live in LDS;
// softmax normalization deferred to epilogue scale. PV over fragment-packed hp.
// ---------------------------------------------------------------------------
__global__ __launch_bounds__(256)
void attn_kernel(const _Float16* __restrict__ HP,      // row-major (A staging)
                 const _Float16* __restrict__ HPPK,    // fragment-packed (B)
                 const _Float16* __restrict__ ASPK,    // fragment-packed Asum
                 const float* __restrict__ bias,
                 float* __restrict__ OUT,
                 int N, int H, int doGelu) {
    extern __shared__ char dynsmem[];
    const int LPh = N + 8;                              // pad vs bank conflicts
    _Float16* logitsH = (_Float16*)dynsmem;             // 16 * LPh f16
    _Float16* hpT     = logitsH + 16 * LPh;             // 16 * 64 f16
    float*    outacc  = (float*)(hpT + 16 * 64);        // 16 * 64 f32
    float*    smInv   = outacc + 16 * 64;               // 16 f32

    const int bh = blockIdx.x, h = bh % H, b = bh / H;
    const int r0 = blockIdx.y * 16;
    const int tid = threadIdx.x, wave = tid >> 5, lane = tid & 31;
    const _Float16* HPb = HP + (size_t)bh * N * 64;

    for (int i = tid; i < 1024; i += blockDim.x) {      // 16x64 contiguous
        hpT[i]    = HPb[(size_t)r0 * 64 + i];
        outacc[i] = 0.f;
    }
    __syncthreads();

    // ---- logits = hp_rows (16x64) @ Asum[h] (64xN), leaky-relu, f16 ----
    const v16h a0 = loadA(hpT + 0,  64, lane);
    const v16h a1 = loadA(hpT + 32, 64, lane);
    const int rowoff = (lane >> 4) << 3, nc = lane & 15;

    for (int jt = wave; jt < N / 16; jt += 8) {
        const _Float16* fb = ASPK + (((size_t)h * (N / 16) + jt) * 2) * 512;
        v8f c = {};
        c = wmma_f16(a0, loadBpk(fb, lane), c);
        c = wmma_f16(a1, loadBpk(fb + 512, lane), c);
#pragma unroll
        for (int r = 0; r < 8; ++r) {
            float v = c[r];
            v = (v > 0.f) ? v : 0.2f * v;               // leaky-relu
            logitsH[(size_t)(r + rowoff) * LPh + jt * 16 + nc] = (_Float16)v;
        }
    }
    __syncthreads();

    // ---- softmax (unnormalized): exp(l - max), keep 1/sum per row ----
#pragma unroll
    for (int rr = 0; rr < 2; ++rr) {
        const int row = wave * 2 + rr;
        _Float16* Lr = logitsH + (size_t)row * LPh;
        float m = -3.0e38f;
        for (int i = lane; i < N; i += 32) m = fmaxf(m, (float)Lr[i]);
#pragma unroll
        for (int off = 16; off > 0; off >>= 1) m = fmaxf(m, __shfl_xor(m, off, 32));
        float s = 0.f;
        for (int i = lane; i < N; i += 32) {
            float e = __expf((float)Lr[i] - m);
            Lr[i] = (_Float16)e;
            s += e;
        }
#pragma unroll
        for (int off = 16; off > 0; off >>= 1) s += __shfl_xor(s, off, 32);
        if (lane == 0) smInv[row] = 1.0f / s;
    }
    __syncthreads();

    // ---- out = P_unnorm (16xN) @ hp (Nx64): K split across 8 waves ----
    v8f acc[4] = {};
    const _Float16* HPKb = HPPK + ((size_t)bh * (N / 32) * 4) * 512;
    for (int kt = wave; kt < N / 32; kt += 8) {
        v16h a = loadA(logitsH + kt * 32, LPh, lane);   // ds_load_b128 x2
        const _Float16* fb = HPKb + (size_t)kt * 4 * 512;
        if (kt + 8 < N / 32)
            __builtin_prefetch((const void*)(fb + 8 * 4 * 512), 0, 1);
#pragma unroll
        for (int t = 0; t < 4; ++t)
            acc[t] = wmma_f16(a, loadBpk(fb + t * 512, lane), acc[t]);
    }
#pragma unroll
    for (int t = 0; t < 4; ++t)
#pragma unroll
        for (int r = 0; r < 8; ++r)
            atomicAdd(&outacc[(size_t)(r + rowoff) * 64 + t * 16 + nc], acc[t][r]);
    __syncthreads();

    // ---- epilogue: row-scale (softmax denom), bias, optional exact GELU ----
    const int outLd = H * 64;
    for (int i = tid; i < 1024; i += blockDim.x) {
        const int row = i >> 6, o = i & 63;
        float v = outacc[i] * smInv[row] + bias[o];
        if (doGelu) v = 0.5f * v * (1.0f + erff(v * 0.70710678118654752f));
        OUT[((size_t)b * N + r0 + row) * outLd + h * 64 + o] = v;
    }
}

// ---------------------------------------------------------------------------
// Host orchestration
// ---------------------------------------------------------------------------
extern "C" void kernel_launch(void* const* d_in, const int* in_sizes, int n_in,
                              void* d_out, int out_size, void* d_ws, size_t ws_size,
                              hipStream_t stream) {
    (void)in_sizes; (void)n_in; (void)out_size; (void)ws_size;
    constexpr int B = 8, N = 2048, H0 = 4, K0 = 64, K1 = 256;

    const float* x      = (const float*)d_in[0];
    const float* w0     = (const float*)d_in[1];
    const float* a_src0 = (const float*)d_in[2];
    const float* a_dst0 = (const float*)d_in[3];
    const float* bias0  = (const float*)d_in[4];
    const float* w1     = (const float*)d_in[5];
    const float* a_src1 = (const float*)d_in[6];
    const float* a_dst1 = (const float*)d_in[7];
    const float* bias1  = (const float*)d_in[8];
    float* out = (float*)d_out;

    char* ws = (char*)d_ws;
    _Float16* xn0   = (_Float16*)(ws + 0);              //  2 MB
    _Float16* hp0   = (_Float16*)(ws + 2097152);        //  8 MB row-major
    _Float16* hppk0 = (_Float16*)(ws + 10485760);       //  8 MB frag-packed
    _Float16* aspk0 = (_Float16*)(ws + 18874368);       //  1 MB
    _Float16* aspk1 = (_Float16*)(ws + 19922944);       // .25 MB
    _Float16* wpk0  = (_Float16*)(ws + 20185088);       // 32 KB
    _Float16* wpk1  = (_Float16*)(ws + 20217856);       // 32 KB
    float*    x1    = (float*)   (ws + 20250624);       // 16 MB
    _Float16* xn1   = (_Float16*)(ws + 37027840);       //  8 MB
    _Float16* hp1   = (_Float16*)(ws + 45416448);       //  2 MB
    _Float16* hppk1 = (_Float16*)(ws + 47513600);       //  2 MB

    const int SMEM_ATTN = 16 * (N + 8) * 2 + 16 * 64 * 2 + 16 * 64 * 4 + 64; // 72000 B
    (void)hipFuncSetAttribute((const void*)attn_kernel,
                              hipFuncAttributeMaxDynamicSharedMemorySize, SMEM_ATTN);

    // ---- one-time operand packing ----
    pack_asum_kernel<<<dim3(128), dim3(256), 0, stream>>>(a_src0, a_dst0, aspk0, H0, N);
    pack_asum_kernel<<<dim3(32),  dim3(256), 0, stream>>>(a_src1, a_dst1, aspk1, 1, N);
    pack_w_kernel<<<dim3(4), dim3(256), 0, stream>>>(w0, wpk0, H0, K0);
    pack_w_kernel<<<dim3(4), dim3(256), 0, stream>>>(w1, wpk1, 1, K1);

    // ---- layer 0 ----
    inorm_kernel<<<dim3(B), dim3(1024), 0, stream>>>(x, xn0, N, 64);
    hpgemm_kernel<<<dim3(B * H0, N / 128), dim3(256), 0, stream>>>(
        xn0, wpk0, hp0, N, K0, H0);
    pack_hp_kernel<<<dim3(1024), dim3(256), 0, stream>>>(hp0, hppk0, B * H0, N);
    attn_kernel<<<dim3(B * H0, N / 16), dim3(256), SMEM_ATTN, stream>>>(
        hp0, hppk0, aspk0, bias0, x1, N, H0, /*gelu=*/1);

    // ---- layer 1 ----
    inorm_kernel<<<dim3(B), dim3(1024), 0, stream>>>(x1, xn1, N, 256);
    hpgemm_kernel<<<dim3(B, N / 128), dim3(256), 0, stream>>>(
        xn1, wpk1, hp1, N, K1, 1);
    pack_hp_kernel<<<dim3(256), dim3(256), 0, stream>>>(hp1, hppk1, B, N);
    attn_kernel<<<dim3(B, N / 16), dim3(256), SMEM_ATTN, stream>>>(
        hp1, hppk1, aspk1, bias1, out, N, 1, /*gelu=*/0);
}
